// Attention_43593918055012
// MI455X (gfx1250) — compile-verified, matrix operand order checked by reading, not run on hardware
//
#include <hip/hip_runtime.h>

// ---------------- problem constants ----------------
#define B_   8
#define N_   1025
#define C_   768
#define H_   12
#define D_   64
#define NP   1040                 // N padded to multiple of 16 (Q/K layout)
#define NPV  1056                 // V^T row stride (multiple of 32, key over-read room)
#define M_   (B_ * N_)            // 8200 flat rows of x
#define MT32 ((M_ + 31) / 32)     // 257 row tiles of 32
#define OC_  (3 * C_)             // 2304 qkv output features
#define SCALE_ 0.125f             // 64^-0.5

typedef __attribute__((ext_vector_type(16))) _Float16 v16h;
typedef __attribute__((ext_vector_type(8)))  float    v8f;

// K index of packed pair j for a given lane-half (ISA 16-bit A/B 16x32 layout):
// half 0 covers K in {0..7, 16..23}, half 1 covers K in {8..15, 24..31}
static __device__ __forceinline__ int kpair(int j, int half) {
  return (j < 4 ? 2 * j : 16 + 2 * (j - 4)) + 8 * half;
}

static __device__ __forceinline__ v8f wmma16(v16h a, v16h b, v8f c) {
  return __builtin_amdgcn_wmma_f32_16x16x32_f16(false, a, false, b, (short)0, c,
                                                false, false);
}

// 16x32 fragment whose 32 K-values are contiguous per lane in memory.
static __device__ __forceinline__ v16h
load_row_frag(const _Float16* __restrict__ p, int row, int ld, int kbase, int half) {
  v16h f;
#pragma unroll
  for (int j = 0; j < 8; ++j) {
    int kk = kpair(j, half);
    const _Float16* q = p + (size_t)row * ld + kbase + kk;
    f[2 * j]     = q[0];
    f[2 * j + 1] = q[1];
  }
  return f;
}

// One ping-pong buffer for the 32x64-tile GEMM inner loop.
struct FragSet { v16h a0, a1, b[4]; };

static __device__ __forceinline__ void
load_set(FragSet& f, const _Float16* __restrict__ A, int arow0, int arow1,
         const _Float16* __restrict__ W, int wrow, int kb, int half) {
  f.a0 = load_row_frag(A, arow0, C_, kb, half);
  f.a1 = load_row_frag(A, arow1, C_, kb, half);
#pragma unroll
  for (int s = 0; s < 4; ++s)
    f.b[s] = load_row_frag(W, wrow + s * 16, C_, kb, half);
}

static __device__ __forceinline__ void compute_set(v8f acc[2][4], const FragSet& f) {
#pragma unroll
  for (int s = 0; s < 4; ++s) {
    acc[0][s] = wmma16(f.a0, f.b[s], acc[0][s]);
    acc[1][s] = wmma16(f.a1, f.b[s], acc[1][s]);
  }
}

// ---------------- stage 0: f32 -> f16 convert ----------------
__global__ void cvt_f32_to_f16(const float* __restrict__ s,
                               _Float16* __restrict__ d, int n) {
  for (int i = blockIdx.x * blockDim.x + threadIdx.x; i < n;
       i += gridDim.x * blockDim.x)
    d[i] = (_Float16)s[i];
}

// ---------------- stage 1: QKV = x @ w_qkv^T -> head-major Q/K and V^T ------
__global__ __launch_bounds__(32) void qkv_gemm(
    const _Float16* __restrict__ xh,   // (M_, 768)
    const _Float16* __restrict__ wh,   // (2304, 768)
    _Float16* __restrict__ qh,         // (B*H, NP, 64), pre-scaled
    _Float16* __restrict__ kh,         // (B*H, NP, 64)
    _Float16* __restrict__ vt) {       // (B*H, 64, NPV)  V transposed
  const int lane = threadIdx.x & 31;
  const int half = lane >> 4;
  const int ln   = lane & 15;
  const int OT   = OC_ / 64;                 // 36
  const int tm   = blockIdx.x / OT;
  const int to   = blockIdx.x % OT;

  int arow0 = tm * 32 + ln;      if (arow0 >= M_) arow0 = M_ - 1;
  int arow1 = tm * 32 + 16 + ln; if (arow1 >= M_) arow1 = M_ - 1;
  const int wrow = to * 64 + ln;

  v8f acc[2][4] = {};
  FragSet f0, f1;                 // ping-pong: loads write sets in place, no copies
  load_set(f0, xh, arow0, arow1, wh, wrow, 0, half);
  for (int kb = 0; kb < C_; kb += 64) {      // C_/32 == 24 chunks (even)
    load_set(f1, xh, arow0, arow1, wh, wrow, kb + 32, half);
    compute_set(acc, f0);
    if (kb + 64 < C_)
      load_set(f0, xh, arow0, arow1, wh, wrow, kb + 64, half);
    compute_set(acc, f1);
  }

#pragma unroll
  for (int t = 0; t < 2; ++t) {
    const int rbase = tm * 32 + t * 16 + 8 * half;   // C/D layout: M = v + 8*half
#pragma unroll
    for (int s = 0; s < 4; ++s) {
      int o     = to * 64 + s * 16 + ln;
      int which = o / C_;
      int rem   = o - which * C_;
      int hh    = rem >> 6;
      int d     = rem & 63;
#pragma unroll
      for (int v = 0; v < 8; ++v) {
        int r = rbase + v;
        if (r >= M_) continue;
        int b = r / N_, n = r - b * N_;
        float val = acc[t][s][v];
        if (which == 0) {
          qh[((size_t)(b * H_ + hh) * NP + n) * D_ + d] = (_Float16)(val * SCALE_);
        } else if (which == 1) {
          kh[((size_t)(b * H_ + hh) * NP + n) * D_ + d] = (_Float16)val;
        } else {
          vt[((size_t)(b * H_ + hh) * D_ + d) * NPV + n] = (_Float16)val;
        }
      }
    }
  }
}

// ---------------- stage 2: flash attention per (b, h, 16-row tile) ----------
__global__ __launch_bounds__(32) void attn_kernel(
    const _Float16* __restrict__ qh, const _Float16* __restrict__ kh,
    const _Float16* __restrict__ vt, _Float16* __restrict__ aout) {
  __shared__ _Float16 pt[16 * 32];           // P staging (one wave per block)

  const int lane = threadIdx.x & 31;
  const int half = lane >> 4;
  const int ln   = lane & 15;
  const int MTA  = NP / 16;                  // 65 row tiles
  const int mt   = blockIdx.x % MTA;
  const int bh   = blockIdx.x / MTA;
  const int b    = bh / H_;
  const int h    = bh % H_;

  const _Float16* Q = qh + (size_t)bh * NP * D_;
  const _Float16* K = kh + (size_t)bh * NP * D_;
  const _Float16* V = vt + (size_t)bh * D_ * NPV;   // V^T: row = dim, col = key

  const int qrow = mt * 16 + ln;
  v16h qa0 = load_row_frag(Q, qrow, D_, 0, half);
  v16h qa1 = load_row_frag(Q, qrow, D_, 32, half);

  v8f acc[4] = {};
  float rmax[8], rsum[8];
#pragma unroll
  for (int v = 0; v < 8; ++v) { rmax[v] = -1e30f; rsum[v] = 0.0f; }

  for (int kc = 0; kc < NP / 32 + 1; ++kc) { // 33 chunks of 32 keys
    const int kb   = kc * 32;
    const int key0 = kb + ln;
    const int key1 = key0 + 16;
    const int ck0  = key0 < NP ? key0 : NP - 1;
    const int ck1  = key1 < NP ? key1 : NP - 1;

    v16h kb00 = load_row_frag(K, ck0, D_, 0, half);
    v16h kb01 = load_row_frag(K, ck0, D_, 32, half);
    v16h kb10 = load_row_frag(K, ck1, D_, 0, half);
    v16h kb11 = load_row_frag(K, ck1, D_, 32, half);
    v8f s0 = v8f{0}; s0 = wmma16(qa0, kb00, s0); s0 = wmma16(qa1, kb01, s0);
    v8f s1 = v8f{0}; s1 = wmma16(qa0, kb10, s1); s1 = wmma16(qa1, kb11, s1);

    // V^T fragments depend only on kb: issue now so softmax VALU hides them.
    v16h vb[4];
#pragma unroll
    for (int s = 0; s < 4; ++s)
      vb[s] = load_row_frag(V, s * 16 + ln, NPV, kb, half);

    const bool ok0 = key0 < N_;
    const bool ok1 = key1 < N_;

    // online softmax; row stats per lane at rows m = v + 8*half (C/D layout)
#pragma unroll
    for (int v = 0; v < 8; ++v) {
      float a0 = ok0 ? s0[v] : -1e30f;
      float a1 = ok1 ? s1[v] : -1e30f;
      float cm = fmaxf(a0, a1);
#pragma unroll
      for (int off = 8; off; off >>= 1)      // 16-lane row reduction (wave32)
        cm = fmaxf(cm, __shfl_xor(cm, off, 32));
      float nm = fmaxf(rmax[v], cm);
      float sc = __expf(rmax[v] - nm);
      rmax[v]  = nm;
      float e0 = ok0 ? __expf(s0[v] - nm) : 0.0f;
      float e1 = ok1 ? __expf(s1[v] - nm) : 0.0f;
      float rs = e0 + e1;
#pragma unroll
      for (int off = 8; off; off >>= 1)
        rs += __shfl_xor(rs, off, 32);
      rsum[v] = rsum[v] * sc + rs;
#pragma unroll
      for (int s = 0; s < 4; ++s) acc[s][v] *= sc;
      const int m = v + 8 * half;
      pt[m * 32 + ln]      = (_Float16)e0;
      pt[m * 32 + ln + 16] = (_Float16)e1;
    }
    __syncthreads();                          // single-wave WG: cheap

    v16h pa;                                  // P as A fragment (row = ln)
#pragma unroll
    for (int j = 0; j < 8; ++j) {
      int kk = kpair(j, half);
      pa[2 * j]     = pt[ln * 32 + kk];
      pa[2 * j + 1] = pt[ln * 32 + kk + 1];
    }
#pragma unroll
    for (int s = 0; s < 4; ++s)
      acc[s] = wmma16(pa, vb[s], acc[s]);
    __syncthreads();
  }

  float inv[8];
#pragma unroll
  for (int v = 0; v < 8; ++v) inv[v] = 1.0f / rsum[v];
  const int rbase = mt * 16 + 8 * half;
#pragma unroll
  for (int s = 0; s < 4; ++s) {
#pragma unroll
    for (int v = 0; v < 8; ++v) {
      int r = rbase + v;
      if (r >= N_) continue;
      aout[(size_t)(b * N_ + r) * C_ + h * D_ + s * 16 + ln] =
          (_Float16)(acc[s][v] * inv[v]);
    }
  }
}

// ---------------- stage 3: out = aout @ w_proj^T + b_proj (fp32 out) --------
__global__ __launch_bounds__(32) void proj_gemm(
    const _Float16* __restrict__ ah,   // (M_, 768)
    const _Float16* __restrict__ wh,   // (768, 768)
    const float* __restrict__ bias,    // (768,)
    float* __restrict__ out) {
  const int lane = threadIdx.x & 31;
  const int half = lane >> 4;
  const int ln   = lane & 15;
  const int OT   = C_ / 64;                  // 12
  const int tm   = blockIdx.x / OT;
  const int to   = blockIdx.x % OT;

  int arow0 = tm * 32 + ln;      if (arow0 >= M_) arow0 = M_ - 1;
  int arow1 = tm * 32 + 16 + ln; if (arow1 >= M_) arow1 = M_ - 1;
  const int wrow = to * 64 + ln;

  v8f acc[2][4] = {};
  FragSet f0, f1;
  load_set(f0, ah, arow0, arow1, wh, wrow, 0, half);
  for (int kb = 0; kb < C_; kb += 64) {
    load_set(f1, ah, arow0, arow1, wh, wrow, kb + 32, half);
    compute_set(acc, f0);
    if (kb + 64 < C_)
      load_set(f0, ah, arow0, arow1, wh, wrow, kb + 64, half);
    compute_set(acc, f1);
  }

#pragma unroll
  for (int t = 0; t < 2; ++t) {
    const int rbase = tm * 32 + t * 16 + 8 * half;
#pragma unroll
    for (int s = 0; s < 4; ++s) {
      int o    = to * 64 + s * 16 + ln;
      float bo = bias[o];
#pragma unroll
      for (int v = 0; v < 8; ++v) {
        int r = rbase + v;
        if (r >= M_) continue;
        out[(size_t)r * C_ + o] = acc[t][s][v] + bo;
      }
    }
  }
}

// ---------------- host launch ----------------
extern "C" void kernel_launch(void* const* d_in, const int* in_sizes, int n_in,
                              void* d_out, int out_size, void* d_ws,
                              size_t ws_size, hipStream_t stream) {
  const float* x      = (const float*)d_in[0];   // (8,1025,768)
  const float* w_qkv  = (const float*)d_in[1];   // (2304,768)
  const float* w_proj = (const float*)d_in[2];   // (768,768)
  const float* b_proj = (const float*)d_in[3];   // (768,)
  float* out = (float*)d_out;

  const size_t nX  = (size_t)M_ * C_;            // 6,297,600
  const size_t nWq = (size_t)OC_ * C_;           // 1,769,472
  const size_t nWp = (size_t)C_ * C_;            // 589,824
  const size_t nQK = (size_t)B_ * H_ * NP * D_;  // 6,389,760
  const size_t nVT = (size_t)B_ * H_ * D_ * NPV; // 6,488,064

  char* ws = (char*)d_ws;
  size_t off = 0;
  auto take = [&](size_t halves) {
    char* p = ws + off;
    off += (halves * sizeof(_Float16) + 255) & ~(size_t)255;
    return (_Float16*)p;
  };
  _Float16* xh  = take(nX);
  _Float16* wqh = take(nWq);
  _Float16* wph = take(nWp);
  _Float16* qh  = take(nQK);
  _Float16* kh  = take(nQK);
  _Float16* vth = take(nVT);
  _Float16* ah  = take(nX);

  cvt_f32_to_f16<<<4096, 256, 0, stream>>>(x, xh, (int)nX);
  cvt_f32_to_f16<<<2048, 256, 0, stream>>>(w_qkv, wqh, (int)nWq);
  cvt_f32_to_f16<<<1024, 256, 0, stream>>>(w_proj, wph, (int)nWp);

  qkv_gemm<<<MT32 * (OC_ / 64), 32, 0, stream>>>(xh, wqh, qh, kh, vth);
  attn_kernel<<<B_ * H_ * (NP / 16), 32, 0, stream>>>(qh, kh, vth, ah);
  proj_gemm<<<MT32 * (C_ / 64), 32, 0, stream>>>(ah, wph, b_proj, out);
}
